// MambaDecoderLayer_59227599012661
// MI455X (gfx1250) — matrix operand order repaired and checked by reading, not compiled, not run on hardware
//
#include <hip/hip_runtime.h>
#include <hip/hip_bf16.h>
#include <cstdint>

typedef unsigned int u32;
typedef unsigned short u16;
typedef unsigned long long u64;
typedef __attribute__((ext_vector_type(2)))  __bf16 v2bf;
typedef __attribute__((ext_vector_type(4)))  u32    v4u;
typedef __attribute__((ext_vector_type(4)))  float  v4f;
typedef __attribute__((ext_vector_type(8)))  float  v8f;
typedef __attribute__((ext_vector_type(16))) __bf16 v16bf;
typedef __attribute__((ext_vector_type(4)))  u32    u32x4;
typedef __attribute__((ext_vector_type(8)))  int    i32x8;
typedef __attribute__((ext_vector_type(4)))  int    i32x4;

union Frag { v4u q[2]; v16bf v; };

#if __has_include(<hip/amd_detail/amd_gfx1250_TDM.h>)
#define TDM_SIX_ARGS 1
#endif
#if __has_builtin(__builtin_amdgcn_tensor_load_to_lds) && __has_builtin(__builtin_amdgcn_s_wait_tensorcnt)
#define USE_TDM 1
#endif

__device__ __forceinline__ u16 f2bf(float a) {
    u32 u = __float_as_uint(a);
    return (u16)((u + 0x7FFFu + ((u >> 16) & 1u)) >> 16); // RNE
}

__device__ __forceinline__ u32 pack_bf2(float a, float b) {
#if __has_builtin(__builtin_amdgcn_cvt_pk_bf16_f32)
    union { v2bf v; u32 u; } cv;
    cv.v = __builtin_amdgcn_cvt_pk_bf16_f32(a, b);
    return cv.u;
#else
    return (u32)f2bf(a) | ((u32)f2bf(b) << 16);
#endif
}

#ifdef USE_TDM
// Issue one TDM 2D tile load: tile_rows x 32 bf16 elements from a row-major
// [rows_total x Kelem] bf16 tensor (row stride ld_elem elements) into LDS at
// byte offset lds_off. D# layout per CDNA5 ISA 8.3/8.4 (count=1, type=2,
// data_size=1 -> 2-byte elements, tile_dim0=32).
__device__ __forceinline__ void tdm_load_2d(
    u32 lds_off, const u32* gptr, int Kelem, int rows_total,
    int tile_rows, int ld_elem)
{
    u64 ga = (u64)(uintptr_t)gptr;
    u32x4 g0;
    g0.x = 1u;                                 // count=1 (valid), user mode
    g0.y = lds_off;                            // lds_addr (bytes)
    g0.z = (u32)ga;                            // global_addr[31:0]
    g0.w = (u32)(ga >> 32) | (2u << 30);       // global_addr[56:32] | type=2
    i32x8 g1;
    g1[0] = 0x00010000;                        // data_size=1 (2B); no pad/mcast
    g1[1] = (int)(((u32)Kelem & 0xFFFFu) << 16);               // tensor_dim0.lo
    g1[2] = (int)(((u32)Kelem >> 16) |
                  (((u32)rows_total & 0xFFFFu) << 16));        // td0.hi|td1.lo
    g1[3] = (int)(((u32)rows_total >> 16) | (32u << 16));      // td1.hi|tile0=32
    g1[4] = tile_rows;                         // tile_dim1 ; tile_dim2=0
    g1[5] = ld_elem;                           // tensor_dim0_stride[31:0]
    g1[6] = 0;                                 // stride0.hi | stride1.lo
    g1[7] = 0;
    i32x4 z4 = (i32x4)0;
#ifdef TDM_SIX_ARGS
    i32x8 z8 = (i32x8)0;
    __builtin_amdgcn_tensor_load_to_lds(g0, g1, z4, z4, z8, 0);
#else
    __builtin_amdgcn_tensor_load_to_lds(g0, g1, z4, z4, 0);
#endif
}
#endif

// ---------------------------------------------------------------------------
// f32 -> packed bf16 pairs (u32). n4 = element_count / 4.
// ---------------------------------------------------------------------------
__global__ __launch_bounds__(256) void pack_bf16(
    const float* __restrict__ in, u32* __restrict__ out, int n4)
{
    int i = blockIdx.x * 256 + threadIdx.x;
    if (i >= n4) return;
    v4f f = *(const v4f*)(in + (size_t)i * 4);
    out[2 * i    ] = pack_bf2(f.x, f.y);
    out[2 * i + 1] = pack_bf2(f.z, f.w);
}

// Extract dt_in slice (128 wide @ offset 10240 of each 10368 row) as bf16.
__global__ __launch_bounds__(256) void dtin_pack(
    const float* __restrict__ zx, u32* __restrict__ out)
{
    int i = blockIdx.x * 256 + threadIdx.x; // 0..65535 (1024 rows x 64 pairs)
    int l = i >> 6, c = i & 63;
    const float* p = zx + (size_t)l * 10368 + 10240 + c * 2;
    out[(size_t)l * 64 + c] = pack_bf2(p[0], p[1]);
}

// ---------------------------------------------------------------------------
// GEMM: C[M,N](f32) = A[M,K](bf16) * W[N,K](bf16)^T, fp32 accumulate.
// A/W given as packed bf16-pair (u32) arrays, K-contiguous; lda2/ldw2 = row
// stride in u32 units (= K/2 for dense). grid = (N/128, M/256), 256 threads.
// Waves 4(M) x 2(N); each wave owns 64x64 = 16 WMMA tiles per K=32 chunk.
// Tile staging: Tensor Data Mover (2 tensor_load_to_lds by wave 0, tracked by
// TENSORcnt) when available; otherwise b128 global->LDS copies.
// ---------------------------------------------------------------------------
__global__ __launch_bounds__(256) void gemm_bf16_wmma(
    const u32* __restrict__ A, int lda2,
    const u32* __restrict__ W, int ldw2,
    float* __restrict__ C, int ldc, int K)
{
    __shared__ u32 lA[256 * 16];  // 256 rows x 32 bf16 = 16 KB
    __shared__ u32 lB[128 * 16];  // 128 rows x 32 bf16 =  8 KB

    const int tid  = threadIdx.x;
    const int lane = tid & 31;
    const int wave = tid >> 5;
    const int wm   = wave >> 1;     // 0..3
    const int wn   = wave & 1;      // 0..1
    const int lm   = lane & 15;
    const int hi   = lane >> 4;     // lane half (0/1)
    const int m0   = blockIdx.y * 256;
    const int n0   = blockIdx.x * 128;

    v8f acc[4][4];
    #pragma unroll
    for (int i = 0; i < 4; ++i)
        #pragma unroll
        for (int j = 0; j < 4; ++j) acc[i][j] = (v8f)0.0f;

    for (int k0 = 0; k0 < K; k0 += 32) {
        const int kp = k0 >> 1; // u32 offset of chunk within row
#ifdef USE_TDM
        if (wave == 0) {
            const int Mtot = (int)gridDim.y * 256;
            const int Ntot = (int)gridDim.x * 128;
            tdm_load_2d((u32)(uintptr_t)&lA[0], A + (size_t)m0 * lda2 + kp,
                        K, Mtot, 256, lda2 * 2);
            tdm_load_2d((u32)(uintptr_t)&lB[0], W + (size_t)n0 * ldw2 + kp,
                        K, Ntot, 128, ldw2 * 2);
            __builtin_amdgcn_s_wait_tensorcnt(0);
        }
        __syncthreads();
#else
        // --- stage A chunk (256 rows x 16 u32) : 1024 b128 copies ---
        #pragma unroll
        for (int i = tid; i < 1024; i += 256) {
            int row = i >> 2, c = i & 3;
            const u32* p = A + (size_t)(m0 + row) * lda2 + kp + c * 4;
            v4u v = *(const v4u*)p;
            if (k0 + 32 < K) __builtin_prefetch(p + 16, 0, 1);
            *(v4u*)&lA[row * 16 + c * 4] = v;
        }
        // --- stage B chunk (128 rows x 16 u32) : 512 b128 copies ---
        #pragma unroll
        for (int i = tid; i < 512; i += 256) {
            int row = i >> 2, c = i & 3;
            const u32* p = W + (size_t)(n0 + row) * ldw2 + kp + c * 4;
            v4u v = *(const v4u*)p;
            if (k0 + 32 < K) __builtin_prefetch(p + 16, 0, 1);
            *(v4u*)&lB[row * 16 + c * 4] = v;
        }
        __syncthreads();
#endif

        // Fragment loads per ISA 7.12.2 (16-bit A 16x32, B 32x16 layouts):
        // A lane<16 holds K pairs {0..7,16..23}, lane>=16 {8..15,24..31}.
        // B uses the same K order as A.
        Frag a[4], b[4];
        const int ka = hi ? 4 : 0;
        const int kb = hi ? 8 : 0;
        #pragma unroll
        for (int mi = 0; mi < 4; ++mi) {
            int m = wm * 64 + mi * 16 + lm;
            a[mi].q[0] = *(const v4u*)&lA[m * 16 + ka];
            a[mi].q[1] = *(const v4u*)&lA[m * 16 + 8 + ka];
        }
        #pragma unroll
        for (int ni = 0; ni < 4; ++ni) {
            int n = wn * 64 + ni * 16 + lm;
            b[ni].q[0] = *(const v4u*)&lB[n * 16 + ka];
            b[ni].q[1] = *(const v4u*)&lB[n * 16 + 8 + ka];
        }
        #pragma unroll
        for (int mi = 0; mi < 4; ++mi)
            #pragma unroll
            for (int ni = 0; ni < 4; ++ni)
                acc[mi][ni] = __builtin_amdgcn_wmma_f32_16x16x32_bf16(
                    false, a[mi].v, false, b[ni].v,
                    (short)0, acc[mi][ni], false, false);
        __syncthreads();
    }

    // C/D layout: VGPR r -> M = r (lanes 0-15) or r+8 (lanes 16-31), N = lane%16
    #pragma unroll
    for (int mi = 0; mi < 4; ++mi) {
        int mbase = m0 + wm * 64 + mi * 16 + hi * 8;
        #pragma unroll
        for (int ni = 0; ni < 4; ++ni) {
            int col = n0 + wn * 64 + ni * 16 + lm;
            #pragma unroll
            for (int r = 0; r < 8; ++r)
                C[(size_t)(mbase + r) * ldc + col] = acc[mi][ni][r];
        }
    }
}

// ---------------------------------------------------------------------------
// RMSNorm: one block per token row; writes bf16 (GEMM operand) directly.
// ---------------------------------------------------------------------------
__global__ __launch_bounds__(256) void rmsnorm_kernel(
    const float* __restrict__ x, const float* __restrict__ w,
    u16* __restrict__ out, int D)
{
    const int row = blockIdx.x, tid = threadIdx.x;
    const float* xr = x + (size_t)row * D;
    float ss = 0.f;
    for (int i = tid; i < D; i += blockDim.x) { float v = xr[i]; ss += v * v; }
    for (int off = 16; off > 0; off >>= 1) ss += __shfl_xor(ss, off, 32);
    __shared__ float red[8];
    __shared__ float s_scale;
    const int lane = tid & 31, wv = tid >> 5, nw = blockDim.x >> 5;
    if (lane == 0) red[wv] = ss;
    __syncthreads();
    if (tid == 0) {
        float t = 0.f;
        for (int i = 0; i < nw; ++i) t += red[i];
        s_scale = rsqrtf(t / (float)D + 1e-5f);
    }
    __syncthreads();
    const float sc = s_scale;
    for (int i = tid; i < D; i += blockDim.x)
        out[(size_t)row * D + i] = f2bf(xr[i] * sc * w[i]);
}

// ---------------------------------------------------------------------------
// dt = softplus(dt_raw + bias)
// ---------------------------------------------------------------------------
__global__ __launch_bounds__(256) void dt_softplus(
    float* __restrict__ dt, const float* __restrict__ bias, int total)
{
    int i = blockIdx.x * 256 + threadIdx.x;
    if (i >= total) return;
    float v = dt[i] + bias[i & 4095];
    dt[i] = (v > 20.0f) ? v : log1pf(__expf(v));
}

// ---------------------------------------------------------------------------
// Selective scan. One lane per (d, n); n = lane % 16, so the 16 state lanes
// of each channel sit in a half-wave -> 4-step shuffle reduction for C.h.
// zx row layout: [z:4096 | xv:1024 | B:1024 | C:4096 | dt_in:128] = 10368.
// Writes yg[l,d] = (y + D*x) * silu(z) as bf16 (feeds out_proj GEMM).
// ---------------------------------------------------------------------------
__global__ __launch_bounds__(256) void mamba_scan(
    const float* __restrict__ zx, const float* __restrict__ dt,
    const float* __restrict__ A_log, const float* __restrict__ D_skip,
    u16* __restrict__ yg)
{
    const int idx = blockIdx.x * 256 + threadIdx.x; // 0..65535
    const int d = idx >> 4;
    const int n = idx & 15;
    const float Acoef = -__expf(A_log[(size_t)d * 16 + n]);
    const float Dv = D_skip[d];
    const int xoff = ((d >> 6) << 4) + (d & 15); // xv[(d/64)*16 + d%16]
    const int boff = ((d >> 6) << 4) + n;        // B [(d/64)*16 + n]
    const int coff = (d & ~15) + n;              // C [(d/16)*16 + n]
    float h = 0.0f;
    for (int l = 0; l < 1024; ++l) {
        const float* row = zx + (size_t)l * 10368;
        float dtv = dt[(size_t)l * 4096 + d];
        float xv  = row[4096 + xoff];
        float Bv  = row[5120 + boff];
        float Cv  = row[6144 + coff];
        float dA  = __expf(dtv * Acoef);
        h = dA * h + dtv * xv * Bv;
        float p = Cv * h;
        p += __shfl_xor(p, 1, 16);
        p += __shfl_xor(p, 2, 16);
        p += __shfl_xor(p, 4, 16);
        p += __shfl_xor(p, 8, 16);
        if (n == 0) {
            float zv = row[d];
            float y  = p + Dv * xv;
            float s  = zv / (1.0f + __expf(-zv));
            yg[(size_t)l * 4096 + d] = f2bf(y * s);
        }
    }
}

// ---------------------------------------------------------------------------
// out_bf16 = silu(g) * u   (feeds down GEMM)
// ---------------------------------------------------------------------------
__global__ __launch_bounds__(256) void silu_mul(
    const float* __restrict__ g, const float* __restrict__ u,
    u16* __restrict__ out, int total)
{
    int i = blockIdx.x * 256 + threadIdx.x;
    if (i >= total) return;
    float x = g[i];
    out[i] = f2bf((x / (1.0f + __expf(-x))) * u[i]);
}

__global__ __launch_bounds__(256) void add_kernel(
    const float* __restrict__ a, const float* __restrict__ b,
    float* __restrict__ c, int total)
{
    int i = blockIdx.x * 256 + threadIdx.x;
    if (i >= total) return;
    c[i] = a[i] + b[i];
}

// ---------------------------------------------------------------------------
extern "C" void kernel_launch(void* const* d_in, const int* in_sizes, int n_in,
                              void* d_out, int out_size, void* d_ws, size_t ws_size,
                              hipStream_t stream)
{
    (void)in_sizes; (void)n_in; (void)out_size; (void)ws_size;
    const float* hidden  = (const float*)d_in[0];
    const float* rms1_w  = (const float*)d_in[1];
    const float* in_proj = (const float*)d_in[2];
    const float* dt_w    = (const float*)d_in[3];
    const float* dt_b    = (const float*)d_in[4];
    const float* A_log   = (const float*)d_in[5];
    const float* D_skip  = (const float*)d_in[6];
    const float* out_w   = (const float*)d_in[7];
    const float* rms2_w  = (const float*)d_in[8];
    const float* gate_w  = (const float*)d_in[9];
    const float* up_w    = (const float*)d_in[10];
    const float* down_w  = (const float*)d_in[11];
    float* out = (float*)d_out;

    const int L = 1024, DM = 2048, DI = 4096, DXB = 1024,
              ZW = 2 * DI + 2 * DXB + 128 /* 10368 */, DTR = 128, NI = 8192;

    // f32 scratch
    float* wsf = (float*)d_ws;
    float* zx  = wsf; wsf += (size_t)L * ZW;   // in_proj out (z|x|B|C|dt_in)
    float* dt  = wsf; wsf += (size_t)L * DI;   // dt raw -> softplus in place
    float* mo  = wsf; wsf += (size_t)L * DM;   // out_proj out
    float* h2  = wsf; wsf += (size_t)L * DM;   // residual 2
    float* g   = wsf; wsf += (size_t)L * NI;   // gate out
    float* u   = wsf; wsf += (size_t)L * NI;   // up out
    float* mlp = wsf; wsf += (size_t)L * DM;   // down out
    // packed bf16 scratch (u32 = 2 bf16)
    u32* wsu   = (u32*)wsf;
    u32* xn16   = wsu; wsu += (size_t)L * DM / 2;        // rmsnorm1 out
    u32* hn16   = wsu; wsu += (size_t)L * DM / 2;        // rmsnorm2 out
    u32* dtin16 = wsu; wsu += (size_t)L * DTR / 2;       // dt_in slice
    u32* yg16   = wsu; wsu += (size_t)L * DI / 2;        // gated scan out
    u32* gact16 = wsu; wsu += (size_t)L * NI / 2;        // silu(g)*u
    u32* wi16   = wsu; wsu += (size_t)ZW * DM / 2;       // in_proj_w
    u32* wd16   = wsu; wsu += (size_t)DI * DTR / 2;      // dt_proj_w
    u32* wo16   = wsu; wsu += (size_t)DM * DI / 2;       // out_proj_w
    u32* wg16   = wsu; wsu += (size_t)NI * DM / 2;       // gate_w
    u32* wu16   = wsu; wsu += (size_t)NI * DM / 2;       // up_w
    u32* wn16   = wsu; wsu += (size_t)DM * NI / 2;       // down_w

    dim3 blk(256);
    // one-time (per call) weight conversion passes: f32 -> packed bf16
    pack_bf16<<<(ZW * DM / 4) / 256, blk, 0, stream>>>(in_proj, wi16, ZW * DM / 4);
    pack_bf16<<<(DI * DTR / 4) / 256, blk, 0, stream>>>(dt_w, wd16, DI * DTR / 4);
    pack_bf16<<<(DM * DI / 4) / 256, blk, 0, stream>>>(out_w, wo16, DM * DI / 4);
    pack_bf16<<<(NI * DM / 4) / 256, blk, 0, stream>>>(gate_w, wg16, NI * DM / 4);
    pack_bf16<<<(NI * DM / 4) / 256, blk, 0, stream>>>(up_w, wu16, NI * DM / 4);
    pack_bf16<<<(DM * NI / 4) / 256, blk, 0, stream>>>(down_w, wn16, DM * NI / 4);

    rmsnorm_kernel<<<L, blk, 0, stream>>>(hidden, rms1_w, (u16*)xn16, DM);
    // zxbcdt = xn @ in_proj^T          [1024 x 2048 x 10368]
    gemm_bf16_wmma<<<dim3(ZW / 128, L / 256), blk, 0, stream>>>(xn16, DM / 2, wi16, DM / 2, zx, ZW, DM);
    // dt_raw = dt_in @ dt_proj^T       [1024 x 128 x 4096]
    dtin_pack<<<(L * DTR / 2) / 256, blk, 0, stream>>>(zx, dtin16);
    gemm_bf16_wmma<<<dim3(DI / 128, L / 256), blk, 0, stream>>>(dtin16, DTR / 2, wd16, DTR / 2, dt, DI, DTR);
    dt_softplus<<<(L * DI) / 256, blk, 0, stream>>>(dt, dt_b, L * DI);
    // sequential selective scan (65536 lanes)
    mamba_scan<<<(DI * 16) / 256, blk, 0, stream>>>(zx, dt, A_log, D_skip, (u16*)yg16);
    // mamba_out = yg @ out_proj^T      [1024 x 4096 x 2048]
    gemm_bf16_wmma<<<dim3(DM / 128, L / 256), blk, 0, stream>>>(yg16, DI / 2, wo16, DI / 2, mo, DM, DI);
    add_kernel<<<(L * DM) / 256, blk, 0, stream>>>(hidden, mo, h2, L * DM);
    rmsnorm_kernel<<<L, blk, 0, stream>>>(h2, rms2_w, (u16*)hn16, DM);
    // gate / up                         [1024 x 2048 x 8192]
    gemm_bf16_wmma<<<dim3(NI / 128, L / 256), blk, 0, stream>>>(hn16, DM / 2, wg16, DM / 2, g, NI, DM);
    gemm_bf16_wmma<<<dim3(NI / 128, L / 256), blk, 0, stream>>>(hn16, DM / 2, wu16, DM / 2, u, NI, DM);
    silu_mul<<<(L * NI) / 256, blk, 0, stream>>>(g, u, (u16*)gact16, L * NI);
    // down                              [1024 x 8192 x 2048]
    gemm_bf16_wmma<<<dim3(DM / 128, L / 256), blk, 0, stream>>>(gact16, NI / 2, wn16, NI / 2, mlp, DM, NI);
    add_kernel<<<(L * DM) / 256, blk, 0, stream>>>(h2, mlp, out, L * DM);
}
